// SGLCModel_nextTimePred_27118423507261
// MI455X (gfx1250) — matrix operand order repaired and checked by reading, not compiled, not run on hardware
//
#include <hip/hip_runtime.h>

typedef __attribute__((ext_vector_type(16))) _Float16 v16h;
typedef __attribute__((ext_vector_type(8)))  float    v8f;
typedef _Float16 h16;

#define T_IN   32
#define T_OUT  32
#define BATCH  64
#define NNODE  64
#define HDIM   128

// ---- workspace (d_ws) layout: f16 transposed weights per cell ----
// per cell (din = 1 or 128):
//   WxruT [256][din], WhruT [256][128], WxcT [128][din], WhcT [128][128]
constexpr int cell_ws_size(int din) {
  return 256 * din + 256 * 128 + 128 * din + 128 * 128;
}
constexpr int OFF_E0 = 0;
constexpr int OFF_E1 = OFF_E0 + cell_ws_size(1);
constexpr int OFF_D0 = OFF_E1 + cell_ws_size(128);
constexpr int OFF_DB = OFF_D0 + cell_ws_size(1);

// ---------------- prep: transpose + convert f32 -> f16 ----------------
// W: [K][N] row-major (f32)  ->  out: [N][K] row-major (f16)
__global__ void k_tr_f32f16(const float* __restrict__ W, h16* __restrict__ out,
                            int K, int N) {
  int i = blockIdx.x * blockDim.x + threadIdx.x;
  if (i >= K * N) return;
  int n = i / K;
  int k = i - n * K;
  out[i] = (h16)W[k * N + n];
}

// ---------------- WMMA fragment helpers (wave32) ----------------
union Frag16 { v16h v; unsigned u[8]; };

// A operand: row-major [m][k], per ISA 16-bit A layout:
// lane half=lane/16, row m=lane&15; VGPR v<4: k=2v+8*half ; v>=4: k=16+2(v-4)+8*half
__device__ __forceinline__ v16h ldA(const h16* p, int ld, int m0, int k0, int lane) {
  Frag16 f;
  const h16* r = p + (m0 + (lane & 15)) * ld + k0 + 8 * (lane >> 4);
  f.u[0] = *(const unsigned*)(r + 0);
  f.u[1] = *(const unsigned*)(r + 2);
  f.u[2] = *(const unsigned*)(r + 4);
  f.u[3] = *(const unsigned*)(r + 6);
  f.u[4] = *(const unsigned*)(r + 16);
  f.u[5] = *(const unsigned*)(r + 18);
  f.u[6] = *(const unsigned*)(r + 20);
  f.u[7] = *(const unsigned*)(r + 22);
  return f.v;
}

// B operand from transposed storage Bt[n][k] (k contiguous):
// n=lane&15 ; VGPR v<4: k=2v+8*half ; v>=4: k=16+2(v-4)+8*half (+k0)
__device__ __forceinline__ v16h ldB(const h16* p, int ld, int n0, int k0, int lane) {
  Frag16 f;
  const h16* r = p + (n0 + (lane & 15)) * ld + k0 + 8 * (lane >> 4);
#pragma unroll
  for (int v = 0; v < 8; ++v) f.u[v] = *(const unsigned*)(r + 2 * v + (v >= 4 ? 8 : 0));
  return f.v;
}

__device__ __forceinline__ v8f wmma16(v16h a, v16h b, v8f c) {
  return __builtin_amdgcn_wmma_f32_16x16x32_f16(false, a, false, b, (short)0, c,
                                                false, false);
}

__device__ __forceinline__ float sigm(float x) { return 1.0f / (1.0f + __expf(-x)); }

// ---------------- one GCGRU cell: h <- GRU(S, x, h) ----------------
// Uses (S*cat)*W == S*(cat*W), and cat*W = x*Wx + h*Wh
// (the x part degenerates to an outer product when DIN==1).
template <int DIN>
__device__ __forceinline__ void gcgru_cell(
    const h16* __restrict__ sS,   // [64][64] LDS, f16
    h16* h,                       // state [64][128] LDS, updated in place
    const h16* xh,                // DIN==128: layer input [64][128] LDS
    const float* xs,              // DIN==1:   scalar input [64] LDS
    const h16* __restrict__ cw,   // cell weights base (global, f16, transposed)
    const float* sbru,            // LDS-staged biases [256]
    const float* sbc,             // LDS-staged biases [128]
    h16* g1t, h16* g2t, h16* rh, h16* ub,
    int lane, int wave) {
  const h16* WxruT = cw;
  const h16* WhruT = WxruT + 256 * DIN;
  const h16* WxcT  = WhruT + 256 * 128;
  const h16* WhcT  = WxcT + 128 * DIN;
  __builtin_prefetch(WhruT, 0, 3);

  // ---- Stage 1: G1 = cat @ Wru   (64 x 256, K = DIN+128), store G1^T ----
#pragma unroll 2
  for (int i = 0; i < 8; ++i) {
    const int t = (i << 3) + wave;  // uniform per wave, constant trip count
    const int mt = (t & 3) << 4;
    const int nt = (t >> 2) << 4;
    v8f c = {};
#pragma unroll
    for (int k0 = 0; k0 < 128; k0 += 32)
      c = wmma16(ldA(h, 128, mt, k0, lane), ldB(WhruT, 128, nt, k0, lane), c);
    if (DIN == 128) {
#pragma unroll
      for (int k0 = 0; k0 < 128; k0 += 32)
        c = wmma16(ldA(xh, 128, mt, k0, lane), ldB(WxruT, 128, nt, k0, lane), c);
    }
    const int n = nt + (lane & 15);
    const float xw = (DIN == 1) ? (float)WxruT[n] : 0.0f;
#pragma unroll
    for (int r = 0; r < 8; ++r) {
      const int m = mt + r + ((lane >> 4) << 3);
      float v = c[r];
      if (DIN == 1) v += xs[m] * xw;
      g1t[n * 64 + m] = (h16)v;  // transposed: G1^T[n][m]
    }
  }
  __syncthreads();

  // ---- Stage 2: Z1 = S @ G1 ; gates r,u = sigmoid(Z1 + bru) ----
#pragma unroll 2
  for (int i = 0; i < 8; ++i) {
    const int t = (i << 3) + wave;
    const int mt = (t & 3) << 4;
    const int nt = (t >> 2) << 4;
    v8f c = {};
    c = wmma16(ldA(sS, 64, mt, 0, lane),  ldB(g1t, 64, nt, 0, lane),  c);
    c = wmma16(ldA(sS, 64, mt, 32, lane), ldB(g1t, 64, nt, 32, lane), c);
    const int n = nt + (lane & 15);
    const float bias = sbru[n];
#pragma unroll
    for (int r = 0; r < 8; ++r) {
      const int m = mt + r + ((lane >> 4) << 3);
      const float g = sigm(c[r] + bias);
      if (n < 128)
        rh[m * 128 + n] = (h16)(g * (float)h[m * 128 + n]);  // r * h
      else
        ub[m * 128 + (n - 128)] = (h16)g;                    // u
    }
  }
  __syncthreads();

  // ---- Stage 3: G2 = cat2 @ Wc   (64 x 128), store G2^T ----
#pragma unroll 2
  for (int i = 0; i < 4; ++i) {
    const int t = (i << 3) + wave;
    const int mt = (t & 3) << 4;
    const int nt = (t >> 2) << 4;
    v8f c = {};
#pragma unroll
    for (int k0 = 0; k0 < 128; k0 += 32)
      c = wmma16(ldA(rh, 128, mt, k0, lane), ldB(WhcT, 128, nt, k0, lane), c);
    if (DIN == 128) {
#pragma unroll
      for (int k0 = 0; k0 < 128; k0 += 32)
        c = wmma16(ldA(xh, 128, mt, k0, lane), ldB(WxcT, 128, nt, k0, lane), c);
    }
    const int n = nt + (lane & 15);
    const float xw = (DIN == 1) ? (float)WxcT[n] : 0.0f;
#pragma unroll
    for (int r = 0; r < 8; ++r) {
      const int m = mt + r + ((lane >> 4) << 3);
      float v = c[r];
      if (DIN == 1) v += xs[m] * xw;
      g2t[n * 64 + m] = (h16)v;
    }
  }
  __syncthreads();

  // ---- Stage 4: Z2 = S @ G2 ; c = tanh ; h = u*h + (1-u)*c ----
#pragma unroll 2
  for (int i = 0; i < 4; ++i) {
    const int t = (i << 3) + wave;
    const int mt = (t & 3) << 4;
    const int nt = (t >> 2) << 4;
    v8f c = {};
    c = wmma16(ldA(sS, 64, mt, 0, lane),  ldB(g2t, 64, nt, 0, lane),  c);
    c = wmma16(ldA(sS, 64, mt, 32, lane), ldB(g2t, 64, nt, 32, lane), c);
    const int n = nt + (lane & 15);
    const float bias = sbc[n];
#pragma unroll
    for (int r = 0; r < 8; ++r) {
      const int m = mt + r + ((lane >> 4) << 3);
      const float cc = tanhf(c[r] + bias);
      const float u = (float)ub[m * 128 + n];
      const float ho = (float)h[m * 128 + n];
      h[m * 128 + n] = (h16)(u * ho + (1.0f - u) * cc);
    }
  }
  __syncthreads();
}

// ---------------- persistent per-batch kernel ----------------
__global__ void __launch_bounds__(256)
k_dcrnn(const float* __restrict__ x_enc,  // [T_IN,B,N,1]
        const float* __restrict__ sup,    // [B,N,N]
        const h16* __restrict__ w,        // transposed f16 weights in d_ws
        const float* __restrict__ bru_e0, const float* __restrict__ bc_e0,
        const float* __restrict__ bru_e1, const float* __restrict__ bc_e1,
        const float* __restrict__ bru_d0, const float* __restrict__ bc_d0,
        const float* __restrict__ bru_db, const float* __restrict__ bc_db,
        const float* __restrict__ Wp, const float* __restrict__ bp,
        float* __restrict__ out) {        // [T_OUT,B,N]
  extern __shared__ char smem_raw[];
  const int tid = threadIdx.x;
  const int lane = tid & 31;
  const int wave = __builtin_amdgcn_readfirstlane(tid >> 5);  // wave-uniform
  const int b = blockIdx.x;

  float* xbuf = (float*)smem_raw;            // 64 f32
  float* bB   = xbuf + 64;                   // 4 cells x (bru 256 | bc 128)
  float* sWp  = bB + 4 * 384;                // 128
  float* sbp  = sWp + 128;                   // 1 (+3 pad)
  h16* sS  = (h16*)(smem_raw + 256 + 6672);  // 64*64
  h16* h0  = sS + 64 * 64;                   // 64*128
  h16* h1  = h0 + 64 * 128;                  // 64*128
  h16* rh  = h1 + 64 * 128;                  // 64*128
  h16* ub  = rh + 64 * 128;                  // 64*128
  h16* g1t = ub + 64 * 128;                  // 256*64
  h16* g2t = g1t + 256 * 64;                 // 128*64

  // init: S -> LDS f16, zero states, stage biases/Wp -> LDS
  for (int i = tid; i < 64 * 64; i += 256) sS[i] = (h16)sup[b * 4096 + i];
  for (int i = tid; i < 64 * 128; i += 256) {
    h0[i] = (h16)0.0f;
    h1[i] = (h16)0.0f;
  }
  for (int i = tid; i < 384; i += 256) {
    bB[0 * 384 + i] = (i < 256) ? bru_e0[i] : bc_e0[i - 256];
    bB[1 * 384 + i] = (i < 256) ? bru_e1[i] : bc_e1[i - 256];
    bB[2 * 384 + i] = (i < 256) ? bru_d0[i] : bc_d0[i - 256];
    bB[3 * 384 + i] = (i < 256) ? bru_db[i] : bc_db[i - 256];
  }
  if (tid < 128) sWp[tid] = Wp[tid];
  if (tid == 0) sbp[0] = bp[0];
  __syncthreads();

  // -------- encoder: 2 stacked cells per timestep --------
  for (int t = 0; t < T_IN; ++t) {
    if (tid < 64) xbuf[tid] = x_enc[(t * BATCH + b) * 64 + tid];
    __syncthreads();
    gcgru_cell<1>(sS, h0, nullptr, xbuf, w + OFF_E0,
                  bB + 0 * 384, bB + 0 * 384 + 256, g1t, g2t, rh, ub, lane, wave);
    gcgru_cell<128>(sS, h1, h0, nullptr, w + OFF_E1,
                    bB + 1 * 384, bB + 1 * 384 + 256, g1t, g2t, rh, ub, lane, wave);
  }

  // -------- decoder: feedback loop --------
  if (tid < 64) xbuf[tid] = 0.0f;  // go symbol
  __syncthreads();
  for (int t = 0; t < T_OUT; ++t) {
    gcgru_cell<1>(sS, h0, nullptr, xbuf, w + OFF_D0,
                  bB + 2 * 384, bB + 2 * 384 + 256, g1t, g2t, rh, ub, lane, wave);
    gcgru_cell<128>(sS, h1, h0, nullptr, w + OFF_DB,
                    bB + 3 * 384, bB + 3 * 384 + 256, g1t, g2t, rh, ub, lane, wave);
    if (tid < 64) {
      float acc = sbp[0];
#pragma unroll 8
      for (int k = 0; k < 128; ++k) acc += (float)h1[tid * 128 + k] * sWp[k];
      out[(t * BATCH + b) * 64 + tid] = acc;
      xbuf[tid] = acc;  // next decoder input
    }
    __syncthreads();
  }
}

// ---------------- host entry ----------------
extern "C" void kernel_launch(void* const* d_in, const int* in_sizes, int n_in,
                              void* d_out, int out_size, void* d_ws, size_t ws_size,
                              hipStream_t stream) {
  (void)in_sizes; (void)n_in; (void)out_size; (void)ws_size;
  const float* x_enc  = (const float*)d_in[0];
  // d_in[1] = targets (unused by forward pass)
  const float* sup    = (const float*)d_in[2];
  const float* e0_Wru = (const float*)d_in[3];
  const float* e0_bru = (const float*)d_in[4];
  const float* e0_Wc  = (const float*)d_in[5];
  const float* e0_bc  = (const float*)d_in[6];
  const float* e1_Wru = (const float*)d_in[7];
  const float* e1_bru = (const float*)d_in[8];
  const float* e1_Wc  = (const float*)d_in[9];
  const float* e1_bc  = (const float*)d_in[10];
  const float* d0_Wru = (const float*)d_in[11];
  const float* d0_bru = (const float*)d_in[12];
  const float* d0_Wc  = (const float*)d_in[13];
  const float* d0_bc  = (const float*)d_in[14];
  const float* db_Wru = (const float*)d_in[15];
  const float* db_bru = (const float*)d_in[16];
  const float* db_Wc  = (const float*)d_in[17];
  const float* db_bc  = (const float*)d_in[18];
  const float* Wp     = (const float*)d_in[19];
  const float* bp     = (const float*)d_in[20];

  h16* w = (h16*)d_ws;

  auto tr = [&](const float* W, h16* o, int K, int N) {
    int tot = K * N;
    hipLaunchKernelGGL(k_tr_f32f16, dim3((tot + 255) / 256), dim3(256), 0,
                       stream, W, o, K, N);
  };
  auto prep = [&](const float* Wru, const float* Wc, h16* base, int din) {
    h16* WxruT = base;
    h16* WhruT = WxruT + 256 * din;
    h16* WxcT  = WhruT + 256 * 128;
    h16* WhcT  = WxcT + 128 * din;
    tr(Wru, WxruT, din, 256);              // x-part of Wru, transposed
    tr(Wru + din * 256, WhruT, 128, 256);  // h-part of Wru, transposed
    tr(Wc, WxcT, din, 128);                // x-part of Wc, transposed
    tr(Wc + din * 128, WhcT, 128, 128);    // h-part of Wc, transposed
  };
  prep(e0_Wru, e0_Wc, w + OFF_E0, 1);
  prep(e1_Wru, e1_Wc, w + OFF_E1, 128);
  prep(d0_Wru, d0_Wc, w + OFF_D0, 1);
  prep(db_Wru, db_Wc, w + OFF_DB, 128);

  const size_t smem = 256 + 6672 +
      sizeof(h16) * (64 * 64 + 4 * 64 * 128 + 256 * 64 + 128 * 64);
  hipLaunchKernelGGL(k_dcrnn, dim3(BATCH), dim3(256), smem, stream,
                     x_enc, sup, w,
                     e0_bru, e0_bc, e1_bru, e1_bc,
                     d0_bru, d0_bc, db_bru, db_bc,
                     Wp, bp, (float*)d_out);
}
